// GATCriticWithLaser_84456236908777
// MI455X (gfx1250) — matrix operand order repaired and checked by reading, not compiled, not run on hardware
//
#include <hip/hip_runtime.h>
#include <hip/hip_bf16.h>

// ---------------------------------------------------------------------------
// GATCriticWithLaser for MI455X (gfx1250): WMMA f16 GEMMs for the two 512x512
// GAT projections and fc1 (wave computes a 16x64 strip: 1 A fragment reused
// across 4 B fragments -> 4 independent v_wmma per K-step); structural
// (complete-K5) attention softmax; scalar kernels for CNN / pools / fc2.
// ---------------------------------------------------------------------------

#define NG    4096
#define NA    5
#define NNODE (NG * NA)      // 20480
#define D     512            // HEADS*HID
#define HEADS 8
#define HID   64

typedef __attribute__((ext_vector_type(16))) _Float16 v16h;
typedef __attribute__((ext_vector_type(8)))  float    v8f;

// ------------------------- fragment K mappings -----------------------------
// 16-bit A 16x32 (CDNA5 ISA): lanes 0-15 -> K {0..7, 16..23},
// lanes 16-31 -> K {8..15, 24..31}; 2 halves per VGPR slot.
__device__ __forceinline__ int frag_k_A(int ks, int lane, int i) {
  int grp = (lane >> 4) & 1;
  return ks * 32 + grp * 8 + (i < 4 ? 2 * i : 16 + 2 * (i - 4));
}
// 16-bit B 32x16: lanes 0-15 -> K {0..7, 16..23},
// lanes 16-31 -> K {8..15, 24..31}; 2 halves per VGPR slot.
__device__ __forceinline__ int frag_k_B(int ks, int lane, int i) {
  int grp = (lane >> 4) & 1;
  return ks * 32 + grp * 8 + (i < 4 ? 2 * i : 16 + 2 * (i - 4));
}

__device__ __forceinline__ unsigned pack2h(float lo, float hi) {
  union { _Float16 h[2]; unsigned u; } p;
  p.h[0] = (_Float16)lo; p.h[1] = (_Float16)hi;
  return p.u;
}

// ------------------------- pack kernels ------------------------------------
// A (M x K row-major, f32) -> fragments [mtile][kstep][lane][8] as u32 pairs.
__global__ void pack_a_f16(const float* __restrict__ A, unsigned* __restrict__ out,
                           int mtiles, int ksteps, int lda, int kact) {
  int idx = blockIdx.x * blockDim.x + threadIdx.x;
  int total = mtiles * ksteps * 256;
  if (idx >= total) return;
  int i    = idx & 7;
  int lane = (idx >> 3) & 31;
  int ks   = (idx >> 8) % ksteps;
  int mt   = (idx >> 8) / ksteps;
  int row  = mt * 16 + (lane & 15);
  int k    = frag_k_A(ks, lane, i);
  float lo = (k     < kact) ? A[(size_t)row * lda + k]     : 0.f;
  float hi = (k + 1 < kact) ? A[(size_t)row * lda + k + 1] : 0.f;
  out[idx] = pack2h(lo, hi);
}

// B (K x N row-major, f32) -> fragments [ntile][kstep][lane][8].
__global__ void pack_b_f16(const float* __restrict__ B, unsigned* __restrict__ out,
                           int ntiles, int ksteps, int ldb, int kact) {
  int idx = blockIdx.x * blockDim.x + threadIdx.x;
  int total = ntiles * ksteps * 256;
  if (idx >= total) return;
  int i    = idx & 7;
  int lane = (idx >> 3) & 31;
  int ks   = (idx >> 8) % ksteps;
  int nt   = (idx >> 8) / ksteps;
  int col  = nt * 16 + (lane & 15);
  int k    = frag_k_B(ks, lane, i);
  float lo = (k     < kact) ? B[(size_t)k       * ldb + col] : 0.f;
  float hi = (k + 1 < kact) ? B[(size_t)(k + 1) * ldb + col] : 0.f;
  out[idx] = pack2h(lo, hi);
}

// comb = [ggnn(512) | glid(5) | gact(3)] padded to K=544, packed directly.
__global__ void pack_comb_f16(const float* __restrict__ ggnn,
                              const float* __restrict__ glid,
                              const float* __restrict__ gact,
                              unsigned* __restrict__ out, int ksteps) {
  int idx = blockIdx.x * blockDim.x + threadIdx.x;
  int total = (NG / 16) * ksteps * 256;
  if (idx >= total) return;
  int i    = idx & 7;
  int lane = (idx >> 3) & 31;
  int ks   = (idx >> 8) % ksteps;
  int mt   = (idx >> 8) / ksteps;
  int g    = mt * 16 + (lane & 15);
  int k    = frag_k_A(ks, lane, i);
  float v[2];
#pragma unroll
  for (int t = 0; t < 2; ++t) {
    int kk = k + t;
    float x = 0.f;
    if (kk < 512)      x = ggnn[(size_t)g * 512 + kk];
    else if (kk < 517) x = glid[(size_t)g * 5 + (kk - 512)];
    else if (kk < 520) x = gact[(size_t)g * 3 + (kk - 517)];
    v[t] = x;
  }
  out[idx] = pack2h(v[0], v[1]);
}

// ------------------------- WMMA GEMM ---------------------------------------
// One wave per 16x64 C strip: one A fragment feeds 4 B fragments -> 4
// independent v_wmma per K-step (A traffic cut 4x, XDL pipe kept busy).
__global__ void __launch_bounds__(32)
wmma_gemm_f16(const unsigned* __restrict__ Apack, const unsigned* __restrict__ Bpack,
              float* __restrict__ C, const float* __restrict__ bias,
              int ksteps, int ldc, int relu) {
  int mt = blockIdx.x, nt4 = blockIdx.y;
  int lane = threadIdx.x;
  const size_t bstride = (size_t)ksteps * 256;   // u32s per packed N-tile
  const unsigned* Af = Apack + ((size_t)mt * ksteps * 32 + lane) * 8;
  const unsigned* Bf = Bpack + (size_t)(nt4 * 4) * bstride + (size_t)lane * 8;
  v8f acc[4] = {};
  for (int ks = 0; ks < ksteps; ++ks) {
    v16h a = *reinterpret_cast<const v16h*>(Af + (size_t)ks * 256);
    if (ks + 1 < ksteps) {
      __builtin_prefetch(Af + (size_t)(ks + 1) * 256, 0, 3);
      __builtin_prefetch(Bf + (size_t)(ks + 1) * 256, 0, 3);
    }
#pragma unroll
    for (int sub = 0; sub < 4; ++sub) {
      v16h b = *reinterpret_cast<const v16h*>(Bf + (size_t)sub * bstride +
                                              (size_t)ks * 256);
      acc[sub] = __builtin_amdgcn_wmma_f32_16x16x32_f16(
          /*neg_a=*/false, a, /*neg_b=*/false, b,
          /*c_mod=*/(short)0, acc[sub], /*reuse_a=*/false, /*reuse_b=*/false);
    }
  }
  // C/D layout: lanes 0-15 -> M=0..7 across VGPRs, lanes 16-31 -> M=8..15.
  int rbase = mt * 16 + ((lane >> 4) & 1) * 8;
#pragma unroll
  for (int sub = 0; sub < 4; ++sub) {
    int col = nt4 * 64 + sub * 16 + (lane & 15);
    float bv = bias ? bias[col] : 0.f;
#pragma unroll
    for (int r = 0; r < 8; ++r) {
      float v = acc[sub][r] + bv;
      if (relu) v = fmaxf(v, 0.f);
      C[(size_t)(rbase + r) * ldc + col] = v;
    }
  }
}

// ------------------------- lidar CNN (one wave per node) -------------------
__global__ void __launch_bounds__(32)
lidar_cnn(const float* __restrict__ x,
          const float* __restrict__ cw1, const float* __restrict__ cb1,
          const float* __restrict__ cw2, const float* __restrict__ cb2,
          const float* __restrict__ lw,  const float* __restrict__ lb,
          float* __restrict__ lid) {
  __shared__ float lbuf[20];
  __shared__ float pool[16][10];
  __shared__ float v[32];
  int n = blockIdx.x, t = threadIdx.x;
  if (t < 20) lbuf[t] = x[(size_t)n * 32 + t];
  __syncthreads();
  // conv1(k=5,pad=2) + relu + maxpool(2) fused: 16 ch x 10 pooled pos
  for (int task = t; task < 160; task += 32) {
    int o = task / 10, q = task % 10;
    float m = -1e30f;
#pragma unroll
    for (int dp = 0; dp < 2; ++dp) {
      int p = 2 * q + dp;
      float a = cb1[o];
#pragma unroll
      for (int tt = 0; tt < 5; ++tt) {
        int ip = p + tt - 2;
        if (ip >= 0 && ip < 20) a += cw1[o * 5 + tt] * lbuf[ip];
      }
      m = fmaxf(m, fmaxf(a, 0.f));
    }
    pool[o][q] = m;
  }
  __syncthreads();
  { // conv2(k=3,pad=1) + relu + mean over positions: one out-channel per lane
    int o = t;
    float sum = 0.f;
    for (int p = 0; p < 10; ++p) {
      float a = cb2[o];
#pragma unroll
      for (int i = 0; i < 16; ++i)
#pragma unroll
        for (int tt = 0; tt < 3; ++tt) {
          int ip = p + tt - 1;
          if (ip >= 0 && ip < 10) a += cw2[(o * 16 + i) * 3 + tt] * pool[i][ip];
        }
      sum += fmaxf(a, 0.f);
    }
    v[o] = sum * 0.1f;
  }
  __syncthreads();
  if (t < 5) {
    float a = lb[t];
#pragma unroll
    for (int i = 0; i < 32; ++i) a += v[i] * lw[i * 5 + t];
    lid[(size_t)n * 5 + t] = fmaxf(a, 0.f);
  }
}

// ------------------------- small GAT helpers -------------------------------
// xp1 = feat(Nx9) @ W1(9x512)
__global__ void gat1_proj(const float* __restrict__ x, const float* __restrict__ W,
                          float* __restrict__ xp) {
  int idx = blockIdx.x * blockDim.x + threadIdx.x;
  if (idx >= NNODE * D) return;
  int n = idx >> 9, c = idx & 511;
  const float* f = x + (size_t)n * 32 + 20;
  float a = 0.f;
#pragma unroll
  for (int k = 0; k < 9; ++k) a += f[k] * W[k * 512 + c];
  xp[idx] = a;
}

// per-node per-head attention dots
__global__ void attn_node(const float* __restrict__ xp,
                          const float* __restrict__ att_src,
                          const float* __restrict__ att_dst,
                          float* __restrict__ a_src, float* __restrict__ a_dst) {
  int idx = blockIdx.x * blockDim.x + threadIdx.x;
  if (idx >= NNODE * HEADS) return;
  int n = idx >> 3, h = idx & 7;
  const float* row = xp + (size_t)n * D + h * HID;
  float s0 = 0.f, s1 = 0.f;
  for (int c = 0; c < HID; ++c) {
    float xv = row[c];
    s0 += xv * att_src[h * HID + c];
    s1 += xv * att_dst[h * HID + c];
  }
  a_src[idx] = s0;
  a_dst[idx] = s1;
}

// s[h] = sum_c We[h*64+c] * att_edge[h*64+c]  (edge_attr is scalar)
__global__ void ae_scale(const float* __restrict__ We,
                         const float* __restrict__ att_edge,
                         float* __restrict__ s) {
  int h = threadIdx.x;
  if (h >= HEADS) return;
  float a = 0.f;
  for (int c = 0; c < HID; ++c) a += We[h * HID + c] * att_edge[h * HID + c];
  s[h] = a;
}

// self-loop attr = mean of the 4 incoming edge attrs of each node
__global__ void loop_attr_kernel(const float* __restrict__ ea,
                                 float* __restrict__ la) {
  int n = blockIdx.x * blockDim.x + threadIdx.x;
  if (n >= NNODE) return;
  int g = n / NA, j = n % NA;
  float s = 0.f;
#pragma unroll
  for (int i = 0; i < NA; ++i) {
    if (i == j) continue;
    int eoff = i * 4 + (j < i ? j : j - 1);
    s += ea[g * 20 + eoff];
  }
  la[n] = s * 0.25f;
}

// dst-side softmax + weighted aggregation + bias + relu (structural K5)
__global__ void __launch_bounds__(128)
gat_aggregate(const float* __restrict__ xp,
              const float* __restrict__ a_src, const float* __restrict__ a_dst,
              const float* __restrict__ ea,    const float* __restrict__ la,
              const float* __restrict__ sae,   const float* __restrict__ bias,
              float* __restrict__ out, int with_self) {
  __shared__ float logit[HEADS][5];
  __shared__ float alpha[HEADS][5];
  __shared__ int   srcs[5];
  int n = blockIdx.x;
  int g = n / NA, j = n % NA;
  int t = threadIdx.x;
  int cnt = with_self ? 5 : 4;
  if (t < 5) {
    int src;
    if (t < 4) { int i = (t < j) ? t : t + 1; src = g * NA + i; }
    else       { src = n; }
    srcs[t] = src;
  }
  if (t < HEADS * 5) {
    int h = t / 5, sidx = t % 5;
    if (sidx < cnt) {
      int src; float attr;
      if (sidx < 4) {
        int i = (sidx < j) ? sidx : sidx + 1;
        src = g * NA + i;
        attr = ea[g * 20 + i * 4 + (j < i ? j : j - 1)];
      } else { src = n; attr = la[n]; }
      float a = a_src[src * HEADS + h] + a_dst[n * HEADS + h] + attr * sae[h];
      logit[h][sidx] = (a > 0.f) ? a : 0.2f * a;   // leaky_relu 0.2
    }
  }
  __syncthreads();
  if (t < HEADS) {
    float m = -1e30f;
    for (int s = 0; s < cnt; ++s) m = fmaxf(m, logit[t][s]);
    float ex[5], den = 0.f;
    for (int s = 0; s < cnt; ++s) { ex[s] = __expf(logit[t][s] - m); den += ex[s]; }
    float inv = 1.f / (den + 1e-16f);
    for (int s = 0; s < cnt; ++s) alpha[t][s] = ex[s] * inv;
  }
  __syncthreads();
  for (int ch = t; ch < D; ch += 128) {
    int h = ch >> 6;
    float acc = 0.f;
    for (int s = 0; s < cnt; ++s)
      acc += alpha[h][s] * xp[(size_t)srcs[s] * D + ch];
    out[(size_t)n * D + ch] = fmaxf(acc + bias[ch], 0.f);
  }
}

// ------------------------- pooling + fc2 -----------------------------------
__global__ void pool_gnn(const float* __restrict__ h, float* __restrict__ ggnn) {
  int idx = blockIdx.x * blockDim.x + threadIdx.x;
  if (idx >= NG * D) return;
  int g = idx >> 9, c = idx & 511;
  float s = 0.f;
#pragma unroll
  for (int a = 0; a < NA; ++a) s += h[(size_t)(g * NA + a) * D + c];
  ggnn[idx] = s * 0.2f;
}
__global__ void pool_lid(const float* __restrict__ lid, float* __restrict__ glid) {
  int idx = blockIdx.x * blockDim.x + threadIdx.x;
  if (idx >= NG * 5) return;
  int g = idx / 5, f = idx % 5;
  float s = 0.f;
#pragma unroll
  for (int a = 0; a < NA; ++a) s += lid[(size_t)(g * NA + a) * 5 + f];
  glid[idx] = s * 0.2f;
}
__global__ void pool_act(const float* __restrict__ x, float* __restrict__ gact) {
  int idx = blockIdx.x * blockDim.x + threadIdx.x;
  if (idx >= NG * 3) return;
  int g = idx / 3, f = idx % 3;
  float s = 0.f;
#pragma unroll
  for (int a = 0; a < NA; ++a) s += x[(size_t)(g * NA + a) * 32 + 29 + f];
  gact[idx] = s * 0.2f;
}
__global__ void fc2_kernel(const float* __restrict__ hid, const float* __restrict__ w,
                           const float* __restrict__ b, float* __restrict__ out) {
  int g = blockIdx.x * blockDim.x + threadIdx.x;
  if (g >= NG) return;
  float a = b[0];
  for (int k = 0; k < 256; ++k) a += hid[(size_t)g * 256 + k] * w[k];
  out[g] = a;
}

// ---------------------------------------------------------------------------
extern "C" void kernel_launch(void* const* d_in, const int* in_sizes, int n_in,
                              void* d_out, int out_size, void* d_ws, size_t ws_size,
                              hipStream_t stream) {
  (void)in_sizes; (void)n_in; (void)out_size; (void)ws_size;
  const float* x    = (const float*)d_in[0];
  const float* ea   = (const float*)d_in[1];
  // d_in[2]=edge_index, d_in[3]=batch_map: structure is known, unused
  const float* cw1  = (const float*)d_in[4];
  const float* cb1  = (const float*)d_in[5];
  const float* cw2  = (const float*)d_in[6];
  const float* cb2  = (const float*)d_in[7];
  const float* lw   = (const float*)d_in[8];
  const float* lb   = (const float*)d_in[9];
  const float* gW[3]   = { (const float*)d_in[10], (const float*)d_in[16], (const float*)d_in[22] };
  const float* gAs[3]  = { (const float*)d_in[11], (const float*)d_in[17], (const float*)d_in[23] };
  const float* gAd[3]  = { (const float*)d_in[12], (const float*)d_in[18], (const float*)d_in[24] };
  const float* gWe[3]  = { (const float*)d_in[13], (const float*)d_in[19], (const float*)d_in[25] };
  const float* gAe[3]  = { (const float*)d_in[14], (const float*)d_in[20], (const float*)d_in[26] };
  const float* gB[3]   = { (const float*)d_in[15], (const float*)d_in[21], (const float*)d_in[27] };
  const float* fc1_w = (const float*)d_in[28];
  const float* fc1_b = (const float*)d_in[29];
  const float* fc2_w = (const float*)d_in[30];
  const float* fc2_b = (const float*)d_in[31];
  float* out = (float*)d_out;

  // ---- workspace layout (256B aligned chunks) ----
  float* ws = (float*)d_ws;
  size_t off = 0;
  auto alloc = [&](size_t nf) { float* p = ws + off; off += (nf + 63) & ~(size_t)63; return p; };
  float* lid   = alloc((size_t)NNODE * 5);
  float* glid  = alloc((size_t)NG * 5);
  float* gact  = alloc((size_t)NG * 3);
  float* la    = alloc(NNODE);
  float* asrc  = alloc((size_t)NNODE * HEADS);
  float* adst  = alloc((size_t)NNODE * HEADS);
  float* sae   = alloc(64);
  float* xp    = alloc((size_t)NNODE * D);
  float* h     = alloc((size_t)NNODE * D);
  float* ggnn  = alloc((size_t)NG * D);
  float* hid   = alloc((size_t)NG * 256);
  unsigned* Apack = (unsigned*)alloc((size_t)(NNODE / 16) * 16 * 256);  // max over uses
  unsigned* Bpack = (unsigned*)alloc((size_t)(D / 16) * 16 * 256);

  const int mtilesN = NNODE / 16;        // 1280
  const int ktD = D / 32;                // 16

  // ---- lidar CNN + pools ----
  lidar_cnn<<<NNODE, 32, 0, stream>>>(x, cw1, cb1, cw2, cb2, lw, lb, lid);
  pool_lid<<<(NG * 5 + 255) / 256, 256, 0, stream>>>(lid, glid);
  pool_act<<<(NG * 3 + 255) / 256, 256, 0, stream>>>(x, gact);
  loop_attr_kernel<<<(NNODE + 255) / 256, 256, 0, stream>>>(ea, la);

  // ---- GAT layer 1 (K=9, scalar projection) ----
  gat1_proj<<<(NNODE * D + 255) / 256, 256, 0, stream>>>(x, gW[0], xp);
  attn_node<<<(NNODE * HEADS + 255) / 256, 256, 0, stream>>>(xp, gAs[0], gAd[0], asrc, adst);
  ae_scale<<<1, 32, 0, stream>>>(gWe[0], gAe[0], sae);
  gat_aggregate<<<NNODE, 128, 0, stream>>>(xp, asrc, adst, ea, la, sae, gB[0], h, 0);

  // ---- GAT layers 2 & 3 (WMMA projections, self-loops) ----
  for (int layer = 1; layer <= 2; ++layer) {
    int na = mtilesN * ktD * 256;
    pack_a_f16<<<(na + 255) / 256, 256, 0, stream>>>(h, Apack, mtilesN, ktD, D, D);
    int nb = (D / 16) * ktD * 256;
    pack_b_f16<<<(nb + 255) / 256, 256, 0, stream>>>(gW[layer], Bpack, D / 16, ktD, D, D);
    wmma_gemm_f16<<<dim3(mtilesN, D / 64), 32, 0, stream>>>(
        Apack, Bpack, xp, nullptr, ktD, D, 0);
    attn_node<<<(NNODE * HEADS + 255) / 256, 256, 0, stream>>>(xp, gAs[layer], gAd[layer], asrc, adst);
    ae_scale<<<1, 32, 0, stream>>>(gWe[layer], gAe[layer], sae);
    gat_aggregate<<<NNODE, 128, 0, stream>>>(xp, asrc, adst, ea, la, sae, gB[layer], h, 1);
  }

  // ---- pools + fc1 (WMMA, K padded 520 -> 544) + fc2 ----
  pool_gnn<<<(NG * D + 255) / 256, 256, 0, stream>>>(h, ggnn);
  const int ktC = 544 / 32;              // 17
  {
    int na = (NG / 16) * ktC * 256;
    pack_comb_f16<<<(na + 255) / 256, 256, 0, stream>>>(ggnn, glid, gact, Apack, ktC);
    int nb = (256 / 16) * ktC * 256;
    pack_b_f16<<<(nb + 255) / 256, 256, 0, stream>>>(fc1_w, Bpack, 256 / 16, ktC, 256, 520);
    wmma_gemm_f16<<<dim3(NG / 16, 256 / 64), 32, 0, stream>>>(
        Apack, Bpack, hid, fc1_b, ktC, 256, 1);
  }
  fc2_kernel<<<(NG + 255) / 256, 256, 0, stream>>>(hid, fc2_w, fc2_b, out);
}
